// AutoregressiveAttention_24902220382338
// MI455X (gfx1250) — matrix operand order repaired and checked by reading, not compiled, not run on hardware
//
#include <hip/hip_runtime.h>

#define DIMC   1024
#define HEADS  16
#define HD     64
#define BATCH  2
#define SEQ    2048
#define MROWS  (BATCH * SEQ)   // 4096
#define NQKV   (3 * DIMC)      // 3072

typedef __attribute__((ext_vector_type(16))) _Float16 v16h;
typedef __attribute__((ext_vector_type(8)))  _Float16 v8h;
typedef __attribute__((ext_vector_type(8)))  float    v8f;

static __device__ __forceinline__ v8f wmma32(v16h a, v16h b, v8f c) {
  // D = A(16x32 f16) * B(32x16 f16) + C(16x16 f32)
  return __builtin_amdgcn_wmma_f32_16x16x32_f16(false, a, false, b, (short)0, c,
                                                false, false);
}

// LDS byte address (wave-relative) of a __shared__ object, for async-to-LDS ops
#define LDS_ADDR(p) \
  ((unsigned)(uintptr_t)(__attribute__((address_space(3))) void*)(p))

// CDNA5 async direct global->LDS copy (16B per lane), tracked by ASYNCcnt
static __device__ __forceinline__ void async_copy_b128(unsigned lds_addr,
                                                       const void* gaddr) {
  unsigned long long ga = (unsigned long long)(uintptr_t)gaddr;
  asm volatile("global_load_async_to_lds_b128 %0, %1, off"
               :: "v"(lds_addr), "v"(ga)
               : "memory");
}

static __device__ __forceinline__ void wait_asynccnt0() {
  asm volatile("s_wait_asynccnt 0x0" ::: "memory");
}

// A-fragment (16x32, f16), row-major source, per ISA layout:
// lanes 0-15: halves 0..7 -> K=0..7, 8..15 -> K=16..23 ; lanes 16-31: +8
static __device__ __forceinline__ v16h load_a_frag(const _Float16* __restrict__ base,
                                                   int lda) {
  const int lane = threadIdx.x & 31;
  const int m  = lane & 15;
  const int kb = (lane >> 4) << 3;   // 0 or 8
  const _Float16* p = base + m * lda + kb;
  v8h lo = *(const v8h*)(p);         // K = kb .. kb+7
  v8h hi = *(const v8h*)(p + 16);    // K = kb+16 .. kb+23
  v16h a;
#pragma unroll
  for (int j = 0; j < 8; ++j) { a[j] = lo[j]; a[j + 8] = hi[j]; }
  return a;
}

// B-fragment (32x16, f16) where element (k,n) = base[n*ld + k]:
// lanes 0-15: n=lane, K=0..7,16..23 ; lanes 16-31: n=lane-16, K=8..15,24..31
static __device__ __forceinline__ v16h load_b_frag(const _Float16* __restrict__ base,
                                                   int ld) {
  const int lane = threadIdx.x & 31;
  const int n  = lane & 15;
  const int kb = (lane >> 4) << 3;   // 0 or 8
  const _Float16* p = base + n * ld + kb;
  v8h lo = *(const v8h*)(p);
  v8h hi = *(const v8h*)(p + 16);
  v16h b;
#pragma unroll
  for (int j = 0; j < 8; ++j) { b[j] = lo[j]; b[j + 8] = hi[j]; }
  return b;
}

// B-fragment read from an LDS tile laid out [n][32]
static __device__ __forceinline__ v16h load_b_frag_lds(const _Float16* tile) {
  const int lane = threadIdx.x & 31;
  const int n  = lane & 15;
  const int kb = (lane >> 4) << 3;
  const _Float16* p = tile + n * 32 + kb;
  v8h lo = *(const v8h*)(p);
  v8h hi = *(const v8h*)(p + 16);
  v16h b;
#pragma unroll
  for (int j = 0; j < 8; ++j) { b[j] = lo[j]; b[j + 8] = hi[j]; }
  return b;
}

// ---------------------------------------------------------------- weights f32->f16
__global__ void cvt_f16_kernel(const float* __restrict__ src,
                               _Float16* __restrict__ dst, int n) {
  int i = blockIdx.x * blockDim.x + threadIdx.x;
  int stride = gridDim.x * blockDim.x;
  for (; i < n; i += stride) dst[i] = (_Float16)src[i];
}

// ---------------------------------------------------------------- LayerNorm -> f16
__global__ __launch_bounds__(256)
void ln_kernel(const float* __restrict__ x, const float* __restrict__ w,
               const float* __restrict__ b, _Float16* __restrict__ xn) {
  __shared__ float red0[8];
  __shared__ float red1[8];
  const int row = blockIdx.x;
  const float* xr = x + (size_t)row * DIMC;
  float s = 0.f, ss = 0.f;
  for (int i = threadIdx.x; i < DIMC; i += 256) {
    float v = xr[i]; s += v; ss += v * v;
  }
#pragma unroll
  for (int off = 16; off; off >>= 1) {
    s  += __shfl_xor(s, off, 32);
    ss += __shfl_xor(ss, off, 32);
  }
  if ((threadIdx.x & 31) == 0) {
    red0[threadIdx.x >> 5] = s;
    red1[threadIdx.x >> 5] = ss;
  }
  __syncthreads();
  float ts = 0.f, tss = 0.f;
#pragma unroll
  for (int i = 0; i < 8; ++i) { ts += red0[i]; tss += red1[i]; }
  const float mu  = ts * (1.0f / DIMC);
  const float var = tss * (1.0f / DIMC) - mu * mu;
  const float inv = rsqrtf(var + 1e-5f);
  for (int i = threadIdx.x; i < DIMC; i += 256)
    xn[(size_t)row * DIMC + i] = (_Float16)(((xr[i] - mu) * inv) * w[i] + b[i]);
}

// ---------------------------------------------------------------- QKV GEMM (WMMA)
// C[4096,3072] = xn @ Wqkv^T + bias ; scatter into Q[bh,t,d], K[bh,t,d], Vt[bh,d,t]
// Double-buffered LDS W tiles filled with async global->LDS copies (ASYNCcnt).
__global__ __launch_bounds__(256)
void qkv_gemm_kernel(const _Float16* __restrict__ xn,
                     const _Float16* __restrict__ wq,
                     const float* __restrict__ bias,
                     _Float16* __restrict__ Q, _Float16* __restrict__ Kc,
                     _Float16* __restrict__ Vt) {
  __shared__ _Float16 wt[2][64 * 32];   // 2 x (64 n-cols x 32 k)
  const int wave = threadIdx.x >> 5;
  const int lane = threadIdx.x & 31;
  const int m0 = blockIdx.x * 128 + wave * 16;
  const int n0 = blockIdx.y * 64;
  const int nn = threadIdx.x >> 2;            // staging: 16B per thread
  const int kk = (threadIdx.x & 3) << 3;
  const _Float16* wrow = wq + (size_t)(n0 + nn) * DIMC + kk;
  v8f acc[4] = {v8f{}, v8f{}, v8f{}, v8f{}};

  // prologue: async-stage k-tile 0 into buffer 0
  async_copy_b128(LDS_ADDR(&wt[0][nn * 32 + kk]), wrow);

  int buf = 0;
  for (int k0 = 0; k0 < DIMC; k0 += 32, buf ^= 1) {
    wait_asynccnt0();     // my chunk of wt[buf] is in LDS
    __syncthreads();      // everyone's chunk is in LDS; prev reads of wt[buf^1] done
    if (k0 + 32 < DIMC)   // prefetch next tile while computing this one
      async_copy_b128(LDS_ADDR(&wt[buf ^ 1][nn * 32 + kk]), wrow + k0 + 32);
    if (k0 + 64 < DIMC)   // warm L2 for the A tile after next
      __builtin_prefetch(xn + (size_t)m0 * DIMC + k0 + 64, 0, 3);

    const v16h a = load_a_frag(xn + (size_t)m0 * DIMC + k0, DIMC);
    v16h bf[4];
#pragma unroll
    for (int nt = 0; nt < 4; ++nt) bf[nt] = load_b_frag_lds(&wt[buf][nt * 16 * 32]);
#pragma unroll
    for (int nt = 0; nt < 4; ++nt) acc[nt] = wmma32(a, bf[nt], acc[nt]);
  }

  // Epilogue: whole block lies inside one (s, h): n0 multiple of 64
  const int s  = n0 >> 10;
  const int h  = (n0 >> 6) & 15;
  const int hi_ = lane >> 4;
  const int nl  = lane & 15;
#pragma unroll
  for (int nt = 0; nt < 4; ++nt) {
    const int d  = nt * 16 + nl;
    const float bv = bias[n0 + nt * 16 + nl];
#pragma unroll
    for (int r = 0; r < 8; ++r) {
      const int row = m0 + r + hi_ * 8;
      const int bb = row >> 11, tt = row & (SEQ - 1);
      const int bh = bb * HEADS + h;
      const _Float16 val = (_Float16)(acc[nt][r] + bv);
      if (s == 0)      Q [(size_t)(bh * SEQ + tt) * HD + d] = val;
      else if (s == 1) Kc[(size_t)(bh * SEQ + tt) * HD + d] = val;
      else             Vt[(size_t)(bh * HD + d) * SEQ + tt] = val;  // transposed
    }
  }
}

// ---------------------------------------------------------------- causal attention
// One wave per 16-query tile; streams keys 32 at a time (flash-style softmax).
__global__ __launch_bounds__(256)
void attn_kernel(const _Float16* __restrict__ Q, const _Float16* __restrict__ Kc,
                 const _Float16* __restrict__ Vt, _Float16* __restrict__ attn) {
  __shared__ _Float16 ptile[8][16 * 32];   // per-wave P transpose scratch
  const int wave = threadIdx.x >> 5;
  const int lane = threadIdx.x & 31;
  const int tile = blockIdx.x * 8 + wave;   // 4096 tiles total
  const int qt = tile & 127;                // T/16 = 128 q-tiles per head
  const int bh = tile >> 7;                 // 0..31
  const int q0 = qt << 4;
  const float scale = 0.125f;               // 1/sqrt(64)
  const int hi_ = lane >> 4;
  const int nl  = lane & 15;

  const _Float16* Qb = Q + (size_t)(bh * SEQ + q0) * HD;
  const v16h qa0 = load_a_frag(Qb, HD);        // d = 0..31
  const v16h qa1 = load_a_frag(Qb + 32, HD);   // d = 32..63

  v8f o0 = v8f{}, o1 = v8f{}, o2 = v8f{}, o3 = v8f{};
  float m_run[8], l_run[8];
#pragma unroll
  for (int r = 0; r < 8; ++r) { m_run[r] = -1e30f; l_run[r] = 0.f; }

  _Float16* pw = &ptile[wave][0];
  const int nkb = (q0 + 47) >> 5;   // key blocks of 32 covering keys 0..q0+15

  for (int kb = 0; kb < nkb; ++kb) {
    const int kk = kb << 5;
    const _Float16* Kb = Kc + (size_t)(bh * SEQ + kk) * HD;
    // batch all 4 K-fragments, then 4 WMMAs back-to-back
    const v16h kb0 = load_b_frag(Kb, HD);
    const v16h kb1 = load_b_frag(Kb + 32, HD);
    const v16h kb2 = load_b_frag(Kb + 16 * HD, HD);
    const v16h kb3 = load_b_frag(Kb + 16 * HD + 32, HD);
    v8f s0 = wmma32(qa0, kb0, v8f{});
    s0 = wmma32(qa1, kb1, s0);
    v8f s1 = wmma32(qa0, kb2, v8f{});
    s1 = wmma32(qa1, kb3, s1);

    // scale + causal mask + online softmax (row reductions within lane halves)
#pragma unroll
    for (int r = 0; r < 8; ++r) {
      const int qrow = q0 + r + hi_ * 8;
      const int key0 = kk + nl, key1 = key0 + 16;
      float a0 = (key0 <= qrow) ? s0[r] * scale : -1e30f;
      float a1 = (key1 <= qrow) ? s1[r] * scale : -1e30f;
      float mx = fmaxf(a0, a1);
#pragma unroll
      for (int off = 8; off; off >>= 1) mx = fmaxf(mx, __shfl_xor(mx, off, 32));
      const float mnew  = fmaxf(m_run[r], mx);
      const float alpha = __expf(m_run[r] - mnew);
      const float p0 = __expf(a0 - mnew);
      const float p1 = __expf(a1 - mnew);
      float rs = p0 + p1;
#pragma unroll
      for (int off = 8; off; off >>= 1) rs += __shfl_xor(rs, off, 32);
      m_run[r] = mnew;
      l_run[r] = l_run[r] * alpha + rs;
      o0[r] *= alpha; o1[r] *= alpha; o2[r] *= alpha; o3[r] *= alpha;
      const int prow = r + hi_ * 8;                       // C-layout row
      pw[prow * 32 + nl]      = (_Float16)p0;             // key col kk+nl
      pw[prow * 32 + 16 + nl] = (_Float16)p1;             // key col kk+16+nl
    }
    // per-wave LDS RAW: DS ops are in-order per wave; wait + compiler barrier
    asm volatile("s_wait_dscnt 0x0" ::: "memory");

    // re-read P (16x32) in A-fragment layout
    v16h pa;
    {
      const _Float16* pp = pw + nl * 32 + (hi_ << 3);
      v8h lo = *(const v8h*)pp;
      v8h hi = *(const v8h*)(pp + 16);
#pragma unroll
      for (int j = 0; j < 8; ++j) { pa[j] = lo[j]; pa[j + 8] = hi[j]; }
    }
    // O += P @ V : V stored transposed [bh, d, t] -> contiguous along keys
    const _Float16* Vb = Vt + (size_t)(bh * HD) * SEQ + kk;
    const v16h vb0 = load_b_frag(Vb + 0 * 16 * SEQ, SEQ);
    const v16h vb1 = load_b_frag(Vb + 1 * 16 * SEQ, SEQ);
    const v16h vb2 = load_b_frag(Vb + 2 * 16 * SEQ, SEQ);
    const v16h vb3 = load_b_frag(Vb + 3 * 16 * SEQ, SEQ);
    o0 = wmma32(pa, vb0, o0);
    o1 = wmma32(pa, vb1, o1);
    o2 = wmma32(pa, vb2, o2);
    o3 = wmma32(pa, vb3, o3);
  }

  // normalize + store f16 into [B*T, DIM] with head offset
  const int b_ = bh >> 4, h_ = bh & 15;
#pragma unroll
  for (int r = 0; r < 8; ++r) {
    const float inv = 1.0f / l_run[r];
    const int t = q0 + r + hi_ * 8;
    _Float16* orow = attn + (size_t)(b_ * SEQ + t) * DIMC + h_ * HD + nl;
    orow[0]  = (_Float16)(o0[r] * inv);
    orow[16] = (_Float16)(o1[r] * inv);
    orow[32] = (_Float16)(o2[r] * inv);
    orow[48] = (_Float16)(o3[r] * inv);
  }
}

// ---------------------------------------------------------------- output proj GEMM
__global__ __launch_bounds__(256)
void proj_gemm_kernel(const _Float16* __restrict__ ain,
                      const _Float16* __restrict__ wp,
                      const float* __restrict__ bias, float* __restrict__ out) {
  __shared__ _Float16 wt[2][64 * 32];
  const int wave = threadIdx.x >> 5;
  const int lane = threadIdx.x & 31;
  const int m0 = blockIdx.x * 128 + wave * 16;
  const int n0 = blockIdx.y * 64;
  const int nn = threadIdx.x >> 2;
  const int kk = (threadIdx.x & 3) << 3;
  const _Float16* wrow = wp + (size_t)(n0 + nn) * DIMC + kk;
  v8f acc[4] = {v8f{}, v8f{}, v8f{}, v8f{}};

  async_copy_b128(LDS_ADDR(&wt[0][nn * 32 + kk]), wrow);

  int buf = 0;
  for (int k0 = 0; k0 < DIMC; k0 += 32, buf ^= 1) {
    wait_asynccnt0();
    __syncthreads();
    if (k0 + 32 < DIMC)
      async_copy_b128(LDS_ADDR(&wt[buf ^ 1][nn * 32 + kk]), wrow + k0 + 32);
    if (k0 + 64 < DIMC)
      __builtin_prefetch(ain + (size_t)m0 * DIMC + k0 + 64, 0, 3);

    const v16h a = load_a_frag(ain + (size_t)m0 * DIMC + k0, DIMC);
    v16h bf[4];
#pragma unroll
    for (int nt = 0; nt < 4; ++nt) bf[nt] = load_b_frag_lds(&wt[buf][nt * 16 * 32]);
#pragma unroll
    for (int nt = 0; nt < 4; ++nt) acc[nt] = wmma32(a, bf[nt], acc[nt]);
  }

  const int hi_ = lane >> 4;
  const int nl  = lane & 15;
#pragma unroll
  for (int nt = 0; nt < 4; ++nt) {
    const int col = n0 + nt * 16 + nl;
    const float bv = bias[col];
#pragma unroll
    for (int r = 0; r < 8; ++r) {
      const int row = m0 + r + hi_ * 8;
      out[(size_t)row * DIMC + col] = acc[nt][r] + bv;
    }
  }
}

// ---------------------------------------------------------------- launcher
extern "C" void kernel_launch(void* const* d_in, const int* in_sizes, int n_in,
                              void* d_out, int out_size, void* d_ws, size_t ws_size,
                              hipStream_t stream) {
  const float* x      = (const float*)d_in[0];
  const float* ln_w   = (const float*)d_in[1];
  const float* ln_b   = (const float*)d_in[2];
  const float* qkv_w  = (const float*)d_in[3];
  const float* qkv_b  = (const float*)d_in[4];
  const float* proj_w = (const float*)d_in[5];
  const float* proj_b = (const float*)d_in[6];
  // d_in[7] = causal mask (implicit in attention kernel)

  char* ws = (char*)d_ws;
  const size_t SZ_XN   = (size_t)MROWS * DIMC * 2;   // 8 MB
  const size_t SZ_QKVW = (size_t)NQKV  * DIMC * 2;   // 6 MB
  const size_t SZ_PRJW = (size_t)DIMC  * DIMC * 2;   // 2 MB
  const size_t SZ_HEAD = (size_t)BATCH * HEADS * SEQ * HD * 2;  // 8 MB each

  _Float16* xn    = (_Float16*)(ws);
  _Float16* wqkv  = (_Float16*)(ws + SZ_XN);
  _Float16* wproj = (_Float16*)(ws + SZ_XN + SZ_QKVW);
  _Float16* Qb    = (_Float16*)(ws + SZ_XN + SZ_QKVW + SZ_PRJW);
  _Float16* Kb    = (_Float16*)(ws + SZ_XN + SZ_QKVW + SZ_PRJW + SZ_HEAD);
  _Float16* Vt    = (_Float16*)(ws + SZ_XN + SZ_QKVW + SZ_PRJW + 2 * SZ_HEAD);
  _Float16* attnb = (_Float16*)(ws + SZ_XN + SZ_QKVW + SZ_PRJW + 3 * SZ_HEAD);

  cvt_f16_kernel<<<512, 256, 0, stream>>>(qkv_w, wqkv, NQKV * DIMC);
  cvt_f16_kernel<<<512, 256, 0, stream>>>(proj_w, wproj, DIMC * DIMC);
  ln_kernel<<<MROWS, 256, 0, stream>>>(x, ln_w, ln_b, xn);
  qkv_gemm_kernel<<<dim3(MROWS / 128, NQKV / 64), 256, 0, stream>>>(
      xn, wqkv, qkv_b, Qb, Kb, Vt);
  attn_kernel<<<(BATCH * HEADS * (SEQ / 16)) / 8, 256, 0, stream>>>(
      Qb, Kb, Vt, attnb);
  proj_gemm_kernel<<<dim3(MROWS / 128, DIMC / 64), 256, 0, stream>>>(
      attnb, wproj, proj_b, (float*)d_out);
}
